// Attention_13675175870787
// MI455X (gfx1250) — matrix operand order repaired and checked by reading, not compiled, not run on hardware
//
#include <hip/hip_runtime.h>
#include <cstdint>

#define SS 16
#define LL 768
#define CC 256
#define HH 8
#define DD 32
#define KCHUNK 32
#define NCHUNK (LL / KCHUNK)   // 24

typedef __attribute__((ext_vector_type(16))) _Float16 v16h;
typedef __attribute__((ext_vector_type(8)))  _Float16 v8h;
typedef __attribute__((ext_vector_type(8)))  float    v8f;
typedef __attribute__((ext_vector_type(4)))  unsigned int u32x4;
typedef __attribute__((ext_vector_type(4)))  int i32x4;
typedef __attribute__((ext_vector_type(8)))  int i32x8;

// TDM builtin is a target builtin: only visible during the device pass.
// Round-3 probe confirmed: device pass has it, TDM header present -> 6 args.
#if defined(__HIP_DEVICE_COMPILE__) && __has_builtin(__builtin_amdgcn_tensor_load_to_lds)
  #define HAVE_TDM 1
  #if __has_include(<hip/amd_detail/amd_gfx1250_TDM.h>)
    #define TDM_ARITY 6
  #else
    #define TDM_ARITY 5
  #endif
#endif

static __device__ __forceinline__ v8f wmma_f16(v16h a, v16h b, v8f c) {
  return __builtin_amdgcn_wmma_f32_16x16x32_f16(false, a, false, b, (short)0, c,
                                                false, false);
}

// ---- fragment loaders (contiguous k order in A and B: k = 16*(lane/16) + element) ----------

// A fragment 16x32 (f16) from row-major f16 source: two contiguous 16B loads.
static __device__ __forceinline__ v16h load_a_f16(const _Float16* __restrict__ src,
                                                  int ld, int row0, int k0, int lane) {
  int r = row0 + (lane & 15);
  int hi = lane >> 4;
  const _Float16* p = src + (size_t)r * ld + k0 + hi * 16;
  v16h a;
#pragma unroll
  for (int j = 0; j < 8; ++j) { a[j] = p[j]; a[8 + j] = p[8 + j]; }
  return a;
}

// B fragment from packed f16: lane's 16 halves contiguous (32B per lane).
// Packed layout: Wp[((nt*8 + kt)*32 + lane)*16 + j], kt=k0/32, nt=n0/16.
static __device__ __forceinline__ v16h load_b_pk(const _Float16* __restrict__ Wp,
                                                 int k0, int n0, int lane) {
  int kt = k0 >> 5, nt = n0 >> 4;
  return *(const v16h*)(Wp + (((size_t)(nt * 8 + kt)) * 32 + lane) * 16);
}

static __device__ __forceinline__ unsigned lds_addr_of(const void* p) {
  // generic pointer to __shared__: low 32 bits are the LDS byte offset
  return (unsigned)(uintptr_t)p;
}

// ---- TDM staging: contiguous 2KB (1024 f16) global -> LDS ------------------
#ifdef HAVE_TDM
static __device__ __forceinline__ void tdm_load_2kb(unsigned lds_addr,
                                                    const _Float16* gsrc) {
  unsigned long long ga = (unsigned long long)(uintptr_t)gsrc;
  // D# group0: count=1, lds_addr, global_addr[56:0], type=2 ("image")
  u32x4 g0;
  g0[0] = 1u;
  g0[1] = lds_addr;
  g0[2] = (unsigned)ga;
  g0[3] = ((unsigned)(ga >> 32) & 0x01FFFFFFu) | (2u << 30);
  // D# group1: data_size=1 (2B), tensor_dim0=1024, tensor_dim1=1,
  //            tile_dim0=1024, tile_dim1=1, tensor_dim0_stride=1024
  i32x8 g1;
  g1[0] = (int)(1u << 16);
  g1[1] = (int)(1024u << 16);
  g1[2] = (int)(1u << 16);
  g1[3] = (int)(1024u << 16);
  g1[4] = 1;
  g1[5] = 1024;
  g1[6] = 0;
  g1[7] = 0;
  i32x4 z4 = {0, 0, 0, 0};
#if TDM_ARITY == 6
  i32x8 z8 = {0, 0, 0, 0, 0, 0, 0, 0};
  __builtin_amdgcn_tensor_load_to_lds(g0, g1, z4, z4, z8, 0);
#else
  __builtin_amdgcn_tensor_load_to_lds(g0, g1, z4, z4, 0);
#endif
}

static __device__ __forceinline__ void wait_tensorcnt0() {
#if __has_builtin(__builtin_amdgcn_s_wait_tensorcnt)
  __builtin_amdgcn_s_wait_tensorcnt(0);
#else
  asm volatile("s_wait_tensorcnt 0" ::: "memory");
#endif
}
#endif  // HAVE_TDM

// ===========================================================================
// Flash attention per (s, h) (placed first for disasm visibility).
// block = 4 waves, each wave owns a 16-row q tile; K/V chunks (32 keys)
// staged in LDS via double-buffered TDM (cooperative-copy fallback).
// ===========================================================================
__global__ __launch_bounds__(128) void attn_kernel(
    const _Float16* __restrict__ Qh, const _Float16* __restrict__ Kh,
    const _Float16* __restrict__ Vpk, const float* __restrict__ bias,
    const unsigned char* __restrict__ kmask, const float* __restrict__ gate,
    _Float16* __restrict__ wa) {
  __shared__ _Float16 kbuf[2][KCHUNK * DD];    // 2 x 2 KB, row-major keys
  __shared__ _Float16 vbuf[2][KCHUNK * DD];    // 2 x 2 KB, B-frag packed
  __shared__ _Float16 pbuf[4][16 * KCHUNK];    // 4 KB (1 KB per wave)

  const int lane = threadIdx.x & 31;
  const int wave = threadIdx.x >> 5;
  const int s = blockIdx.z, h = blockIdx.y;
  const int qBase = (blockIdx.x * 4 + wave) * 16;

  const _Float16* Qp = Qh + ((size_t)(s * HH + h) * LL) * DD;
  const _Float16* Kp = Kh + ((size_t)(s * HH + h) * LL) * DD;
  const _Float16* Vp = Vpk + ((size_t)(s * HH + h) * LL) * DD;  // packed tiles
  const float* biasp = bias + (size_t)h * LL * LL;
  const unsigned char* maskp = kmask + (size_t)s * LL;

  const int r = lane & 15, hi = lane >> 4;

  v16h aq = load_a_f16(Qp, DD, qBase, 0, lane);

  float m[8], sum[8];
#pragma unroll
  for (int g = 0; g < 8; ++g) { m[g] = -1e30f; sum[g] = 0.0f; }
  v8f o0 = {}, o1 = {};

#ifdef HAVE_TDM
  // prologue: kick off chunk 0 into buffer 0
  if (wave == 0) {
    tdm_load_2kb(lds_addr_of(&kbuf[0][0]), Kp);
    tdm_load_2kb(lds_addr_of(&vbuf[0][0]), Vp);
  }
#endif

  int p = 0;
  for (int kb = 0; kb < LL; kb += KCHUNK) {
#ifdef HAVE_TDM
    if (wave == 0) wait_tensorcnt0();   // chunk kb has landed
    __syncthreads();
    // issue next chunk into the other buffer while everyone computes
    if (wave == 0 && kb + KCHUNK < LL) {
      tdm_load_2kb(lds_addr_of(&kbuf[p ^ 1][0]), Kp + (size_t)(kb + KCHUNK) * DD);
      tdm_load_2kb(lds_addr_of(&vbuf[p ^ 1][0]), Vp + (size_t)(kb + KCHUNK) * DD);
    }
#else
    {
      int t = threadIdx.x;  // 128 threads x 16B = 2 KB each
      ((v8h*)kbuf[p])[t] = ((const v8h*)(Kp + (size_t)kb * DD))[t];
      ((v8h*)vbuf[p])[t] = ((const v8h*)(Vp + (size_t)kb * DD))[t];
    }
    __syncthreads();
#endif

    if (kb + KCHUNK < LL)
      __builtin_prefetch(&biasp[(size_t)(qBase + r) * LL + kb + KCHUNK], 0, 0);

    // ---- logits: Q(16x32) x K^T(32x16) per 16-key tile -> 2 WMMAs ----
    v8f c0 = {}, c1 = {};
    {
      // B operand for K^T: lane reads K[key = t*16 + n][hi*16 .. +15] (contiguous)
      const v16h* bk0 = (const v16h*)&kbuf[p][(size_t)(0 * 16 + r) * DD + hi * 16];
      const v16h* bk1 = (const v16h*)&kbuf[p][(size_t)(1 * 16 + r) * DD + hi * 16];
      c0 = wmma_f16(aq, *bk0, c0);
      c1 = wmma_f16(aq, *bk1, c1);
    }

    // ---- bias + mask ----
#pragma unroll
    for (int g = 0; g < 8; ++g) {
      int row = qBase + g + hi * 8;
      int col0 = kb + r, col1 = col0 + 16;
      c0[g] += biasp[(size_t)row * LL + col0];
      c1[g] += biasp[(size_t)row * LL + col1];
      if (!maskp[col0]) c0[g] = -1e30f;
      if (!maskp[col1]) c1[g] = -1e30f;
    }

    // ---- online softmax (row lives in a 16-lane half) ----
    _Float16* pt = pbuf[wave];
#pragma unroll
    for (int g = 0; g < 8; ++g) {
      float mx = fmaxf(c0[g], c1[g]);
#pragma unroll
      for (int off = 1; off < 16; off <<= 1) mx = fmaxf(mx, __shfl_xor(mx, off, 32));
      float mnew = fmaxf(m[g], mx);
      float scl = __expf(m[g] - mnew);
      float p0 = __expf(c0[g] - mnew);
      float p1 = __expf(c1[g] - mnew);
      float ps = p0 + p1;
#pragma unroll
      for (int off = 1; off < 16; off <<= 1) ps += __shfl_xor(ps, off, 32);
      sum[g] = sum[g] * scl + ps;
      m[g] = mnew;
      o0[g] *= scl;
      o1[g] *= scl;
      int prow = g + hi * 8;
      pt[prow * KCHUNK + r]      = (_Float16)p0;
      pt[prow * KCHUNK + 16 + r] = (_Float16)p1;
    }
    asm volatile("" ::: "memory");  // order P stores before A-frag reloads

    // ---- O += P(16x32) x V(32x32): LDS A-frag + packed V B-frags ----
    {
      v16h ap;
      const _Float16* pr = pt + r * KCHUNK + hi * 16;
#pragma unroll
      for (int j = 0; j < 8; ++j) { ap[j] = pr[j]; ap[8 + j] = pr[8 + j]; }
      const v16h* bv0 = (const v16h*)&vbuf[p][(size_t)lane * 16];
      const v16h* bv1 = (const v16h*)&vbuf[p][(size_t)(32 + lane) * 16];
      o0 = wmma_f16(ap, *bv0, o0);
      o1 = wmma_f16(ap, *bv1, o1);
    }
    __syncthreads();  // all waves done with buffer p before it is refilled
    p ^= 1;
  }

  // ---- epilogue: normalize, gate, store merged-head f16 ----
  const float* gp = gate + (size_t)(s * LL) * CC;
  _Float16* wap = wa + (size_t)(s * LL) * CC;
#pragma unroll
  for (int g = 0; g < 8; ++g) {
    int row = qBase + g + hi * 8;
    int c0i = h * DD + r, c1i = c0i + 16;
    float inv = 1.0f / sum[g];
    wap[(size_t)row * CC + c0i] = (_Float16)(o0[g] * inv * gp[(size_t)row * CC + c0i]);
    wap[(size_t)row * CC + c1i] = (_Float16)(o1[g] * inv * gp[(size_t)row * CC + c1i]);
  }
}

// ===========================================================================
// Prep kernel A: convert q_data/k_data f32 -> f16 row-major.
// ===========================================================================
__global__ __launch_bounds__(256) void cvt_input_kernel(
    const float* __restrict__ q, const float* __restrict__ k,
    _Float16* __restrict__ qh, _Float16* __restrict__ kh) {
  size_t i = (size_t)blockIdx.x * CC + threadIdx.x;
  if (blockIdx.y == 0) qh[i] = (_Float16)q[i];
  else                 kh[i] = (_Float16)k[i];
}

// ===========================================================================
// Prep kernel B: pack 5 weight matrices into B-fragment-native f16 layout.
// ===========================================================================
__global__ __launch_bounds__(256) void pack_w_kernel(
    const float* __restrict__ Wq, const float* __restrict__ Wk,
    const float* __restrict__ Wv, const float* __restrict__ Wg,
    const float* __restrict__ Wo, _Float16* __restrict__ packed) {
  int idx = blockIdx.x * 256 + threadIdx.x;   // element within 256x256
  int w = blockIdx.y;
  const float* W = (w == 0) ? Wq : (w == 1) ? Wk : (w == 2) ? Wv
                 : (w == 3) ? Wg : Wo;
  int k = idx >> 8, n = idx & 255;
  int kt = k >> 5, hi = (k >> 4) & 1, nt = n >> 4, nl = n & 15, j = k & 15;
  int lane = hi * 16 + nl;
  _Float16* dst = packed + (size_t)w * CC * CC;
  dst[(((size_t)(nt * 8 + kt)) * 32 + lane) * 16 + j] = (_Float16)W[idx];
}

// ===========================================================================
// Fused projections.  blockIdx.y: 0=Q 1=K 2=V 3=Gate
// V is written in B-fragment-packed per-(s,h) layout for the PV WMMA.
// ===========================================================================
__global__ __launch_bounds__(128) void proj_kernel(
    const _Float16* __restrict__ xq, const _Float16* __restrict__ xk,
    const _Float16* __restrict__ wpk, const float* __restrict__ bg,
    _Float16* __restrict__ Qh, _Float16* __restrict__ Kh,
    _Float16* __restrict__ Vpk, float* __restrict__ gate) {
  const int lane = threadIdx.x & 31;
  const int wave = threadIdx.x >> 5;
  const int rowBase = blockIdx.x * 16;   // row in [0, S*L)
  const int proj = blockIdx.y;
  const _Float16* X = (proj == 1 || proj == 2) ? xk : xq;
  const _Float16* W = wpk + (size_t)proj * CC * CC;
  const int nBase = wave * 64;

  v8f acc[4] = {};
  for (int k0 = 0; k0 < CC; k0 += 32) {
    v16h a = load_a_f16(X, CC, rowBase, k0, lane);
#pragma unroll
    for (int t = 0; t < 4; ++t) {
      v16h b = load_b_pk(W, k0, nBase + t * 16, lane);
      acc[t] = wmma_f16(a, b, acc[t]);
    }
  }

  const int hi = lane >> 4;
#pragma unroll
  for (int t = 0; t < 4; ++t) {
    int col = nBase + t * 16 + (lane & 15);
#pragma unroll
    for (int g = 0; g < 8; ++g) {
      int row = rowBase + g + hi * 8;     // global row = s*L + l
      float y = acc[t][g];
      if (proj == 3) {
        gate[(size_t)row * CC + col] = 1.0f / (1.0f + __expf(-(y + bg[col])));
      } else {
        int s = row / LL, l = row % LL;
        int h = col >> 5, dd = col & 31;
        if (proj == 2) {
          // B-frag packed V: tile (chunk c = l/32, coltile tt = dd/16)
          int c = l >> 5, khi = (l >> 4) & 1, j = l & 15;
          int tt = dd >> 4, vlane = khi * 16 + (dd & 15);
          Vpk[((((((size_t)s * HH + h) * NCHUNK + c) * 2 + tt) * 32) + vlane) * 16 + j] =
              (_Float16)y;
        } else {
          _Float16 v = (_Float16)(proj == 0 ? y * 0.17677669529663689f : y);
          _Float16* dst = (proj == 0) ? Qh : Kh;
          dst[(((size_t)s * HH + h) * LL + l) * DD + dd] = v;
        }
      }
    }
  }
}

// ===========================================================================
// Output projection  wa(f16) @ Wo(packed f16) + bo -> f32 out
// ===========================================================================
__global__ __launch_bounds__(128) void oproj_kernel(
    const _Float16* __restrict__ wa, const _Float16* __restrict__ wopk,
    const float* __restrict__ bo, float* __restrict__ out) {
  const int lane = threadIdx.x & 31;
  const int wave = threadIdx.x >> 5;
  const int rowBase = blockIdx.x * 16;
  const int nBase = wave * 64;

  v8f acc[4] = {};
  for (int k0 = 0; k0 < CC; k0 += 32) {
    v16h a = load_a_f16(wa, CC, rowBase, k0, lane);
#pragma unroll
    for (int t = 0; t < 4; ++t) {
      v16h b = load_b_pk(wopk, k0, nBase + t * 16, lane);
      acc[t] = wmma_f16(a, b, acc[t]);
    }
  }
  const int hi = lane >> 4;
#pragma unroll
  for (int t = 0; t < 4; ++t) {
    int col = nBase + t * 16 + (lane & 15);
#pragma unroll
    for (int g = 0; g < 8; ++g) {
      int row = rowBase + g + hi * 8;
      out[(size_t)row * CC + col] = acc[t][g] + bo[col];
    }
  }
}

// ===========================================================================
extern "C" void kernel_launch(void* const* d_in, const int* in_sizes, int n_in,
                              void* d_out, int out_size, void* d_ws, size_t ws_size,
                              hipStream_t stream) {
  (void)in_sizes; (void)n_in; (void)out_size; (void)ws_size;
  const float* q_data = (const float*)d_in[0];
  const float* k_data = (const float*)d_in[1];
  const float* bias   = (const float*)d_in[2];
  const unsigned char* kmask = (const unsigned char*)d_in[3];
  const float* Wq = (const float*)d_in[4];
  const float* Wk = (const float*)d_in[5];
  const float* Wv = (const float*)d_in[6];
  const float* Wg = (const float*)d_in[7];
  const float* bg = (const float*)d_in[8];
  const float* Wo = (const float*)d_in[9];
  const float* bo = (const float*)d_in[10];
  float* out = (float*)d_out;

  // workspace layout (all offsets 256B-aligned)
  char* ws = (char*)d_ws;
  const size_t qkvBytes  = (size_t)SS * HH * LL * DD * sizeof(_Float16);  // 6,291,456
  const size_t gateBytes = (size_t)SS * LL * CC * sizeof(float);          // 12,582,912
  const size_t waBytes   = (size_t)SS * LL * CC * sizeof(_Float16);       // 6,291,456
  const size_t xBytes    = (size_t)SS * LL * CC * sizeof(_Float16);       // 6,291,456
  _Float16* Qh  = (_Float16*)(ws);
  _Float16* Kh  = (_Float16*)(ws + qkvBytes);
  _Float16* Vpk = (_Float16*)(ws + 2 * qkvBytes);
  float*    gt  = (float*)(ws + 3 * qkvBytes);
  _Float16* waB = (_Float16*)(ws + 3 * qkvBytes + gateBytes);
  _Float16* Xqh = (_Float16*)(ws + 3 * qkvBytes + gateBytes + waBytes);
  _Float16* Xkh = (_Float16*)(ws + 3 * qkvBytes + gateBytes + waBytes + xBytes);
  _Float16* Wpk = (_Float16*)(ws + 3 * qkvBytes + gateBytes + waBytes + 2 * xBytes);

  dim3 blk128(128), blk256(256);
  cvt_input_kernel<<<dim3(SS * LL, 2), blk256, 0, stream>>>(q_data, k_data, Xqh, Xkh);
  pack_w_kernel<<<dim3(256, 5), blk256, 0, stream>>>(Wq, Wk, Wv, Wg, Wo, Wpk);
  proj_kernel<<<dim3(SS * LL / 16, 4), blk128, 0, stream>>>(
      Xqh, Xkh, Wpk, bg, Qh, Kh, Vpk, gt);
  attn_kernel<<<dim3(LL / 64, HH, SS), blk128, 0, stream>>>(
      Qh, Kh, Vpk, bias, kmask, gt, waB);
  oproj_kernel<<<dim3(SS * LL / 16), blk128, 0, stream>>>(waB, Wpk + 4 * (size_t)CC * CC, bo, out);
}
